// SoftSortOperator_43069932044622
// MI455X (gfx1250) — matrix operand order repaired and checked by reading, not compiled, not run on hardware
//
#include <hip/hip_runtime.h>

// SoftSort / Sinkhorn operator for MI455X (gfx1250).
// One workgroup per batch; the full 256x256 f32 log_alpha matrix lives in LDS
// (268 KB of the 320 KB/WGP). Pad stride 260 (mod 64 == 4):
//  - row sweeps use float4 (ds_load_b128/ds_store_b128): each wave instruction
//    covers all 64 banks exactly twice -> full LDS bandwidth, 4x fewer DS ops
//  - column sweeps (bank = 4i + t) are conflict-free b32
// Sinkhorn runs entirely out of LDS; HBM traffic is ~6 MB total instead of
// ~40 GB for a materialized [2048,256,256] tensor.
// Final perm @ x is executed on the matrix unit: v_wmma_f32_16x16x32_f16.

#define N_ITERS 20
#define NN 256
#define STRIDE 260        // 260 % 64 == 4; rows 16B-aligned for b128 sweeps
#define TAU_INV 10.0f

typedef __attribute__((ext_vector_type(16))) _Float16 v16h;
typedef __attribute__((ext_vector_type(8)))  float    v8f;

extern "C" __global__ __launch_bounds__(256)
void softsort_sinkhorn_kernel(const float* __restrict__ x,
                              const float* __restrict__ x_norm,
                              const float* __restrict__ ref,
                              float* __restrict__ out)
{
    extern __shared__ float smem[];
    float* la   = smem;                 // NN * STRIDE floats (266,240 B)
    float* refs = smem + NN * STRIDE;   // NN floats (16B-aligned offset)
    float* xs   = refs + NN;            // NN floats (16B-aligned offset)

    const int b = blockIdx.x;
    const int t = threadIdx.x;          // 0..255

    refs[t] = ref[t];
    xs[t]   = x[b * NN + t];
    const float xn = x_norm[b * NN + t];
    __syncthreads();

    float4* row4 = (float4*)(la + t * STRIDE);   // thread t's row, 16B-aligned

    // Initialize log_alpha row t: la[t][j] = -|xn - ref[j]| / tau
    {
        const float4* r4 = (const float4*)refs;
        #pragma unroll 8
        for (int q = 0; q < NN / 4; ++q) {
            const float4 r = r4[q];
            float4 o;
            o.x = -fabsf(xn - r.x) * TAU_INV;
            o.y = -fabsf(xn - r.y) * TAU_INV;
            o.z = -fabsf(xn - r.z) * TAU_INV;
            o.w = -fabsf(xn - r.w) * TAU_INV;
            row4[q] = o;
        }
    }
    __syncthreads();

    // 20 Sinkhorn iterations, fully LDS-resident.
    for (int it = 0; it < N_ITERS; ++it) {
        // ---- row logsumexp-normalize: thread t owns row t (float4 sweeps) ----
        float m = -3.402823466e38f;
        #pragma unroll 8
        for (int q = 0; q < NN / 4; ++q) {
            const float4 v = row4[q];
            m = fmaxf(m, fmaxf(fmaxf(v.x, v.y), fmaxf(v.z, v.w)));
        }
        float s = 0.0f;
        #pragma unroll 8
        for (int q = 0; q < NN / 4; ++q) {
            const float4 v = row4[q];
            s += __expf(v.x - m) + __expf(v.y - m) +
                 __expf(v.z - m) + __expf(v.w - m);
        }
        float lse = m + __logf(s);
        #pragma unroll 8
        for (int q = 0; q < NN / 4; ++q) {
            float4 v = row4[q];
            v.x -= lse; v.y -= lse; v.z -= lse; v.w -= lse;
            row4[q] = v;
        }
        __syncthreads();

        // ---- column logsumexp-normalize: thread t owns column t ----
        m = -3.402823466e38f;
        #pragma unroll 8
        for (int i = 0; i < NN; ++i) m = fmaxf(m, la[i * STRIDE + t]);
        s = 0.0f;
        #pragma unroll 8
        for (int i = 0; i < NN; ++i) s += __expf(la[i * STRIDE + t] - m);
        lse = m + __logf(s);
        #pragma unroll 8
        for (int i = 0; i < NN; ++i) la[i * STRIDE + t] -= lse;
        __syncthreads();
    }

    // ---- out[b,i] = sum_j exp(la[i,j]) * x[b,j] via WMMA f16 16x16x32 ----
    // Each wave owns two 16-row tiles; B operand = x chunk replicated across
    // all 16 N-columns, so every D column holds the mat-vec partial sums.
    const int wave = t >> 5;            // 0..7 (uniform per wave)
    const int lane = t & 31;
    const int half = lane >> 4;         // 0: lanes 0-15, 1: lanes 16-31
    const int ml   = lane & 15;         // M within tile (A) / N column (B,D)

    for (int tile = wave; tile < 16; tile += 8) {
        const int i0  = tile * 16;
        const int row = i0 + ml;
        v8f acc = {};
        #pragma unroll
        for (int ks = 0; ks < 8; ++ks) {
            const int j0 = ks * 32;
            // A tile 16x32 f16: lanes 0-15 hold K 0..7 & 16..23,
            //                   lanes 16-31 hold K 8..15 & 24..31.
            const int kb = half * 8;
            const float4* lr  = (const float4*)(la + row * STRIDE + j0 + kb);
            const float4* lr2 = (const float4*)(la + row * STRIDE + j0 + 16 + kb);
            float avals[16];
            *(float4*)(avals + 0)  = lr[0];
            *(float4*)(avals + 4)  = lr[1];
            *(float4*)(avals + 8)  = lr2[0];
            *(float4*)(avals + 12) = lr2[1];
            v16h av;
            #pragma unroll
            for (int e = 0; e < 16; ++e)
                av[e] = (_Float16)__expf(avals[e]);

            // B tile 32x16 f16: lanes 0-15 hold K 0..7 & 16..23, lanes 16-31 K 8..15 & 24..31;
            // x chunk replicated across all N columns.
            const float4* xp = (const float4*)(xs + j0 + half * 8);
            float bvals[16];
            *(float4*)(bvals + 0)  = xp[0];
            *(float4*)(bvals + 4)  = xp[1];
            *(float4*)(bvals + 8)  = xp[4];
            *(float4*)(bvals + 12) = xp[5];
            v16h bv;
            #pragma unroll
            for (int e = 0; e < 16; ++e)
                bv[e] = (_Float16)bvals[e];

            acc = __builtin_amdgcn_wmma_f32_16x16x32_f16(
                false, av, false, bv, (short)0, acc, false, false);
        }
        // D layout: lanes 0-15: M=r, N=lane; lanes 16-31: M=8+r, N=lane-16.
        // All N columns are identical; lanes with N==0 write 8 rows each.
        if (ml == 0) {
            const int mbase = i0 + half * 8;
            #pragma unroll
            for (int r = 0; r < 8; ++r)
                out[b * NN + mbase + r] = acc[r];
        }
    }
}

extern "C" void kernel_launch(void* const* d_in, const int* in_sizes, int n_in,
                              void* d_out, int out_size, void* d_ws, size_t ws_size,
                              hipStream_t stream) {
    const float* x      = (const float*)d_in[0];   // [B, N]
    const float* x_norm = (const float*)d_in[1];   // [B, N]
    const float* ref    = (const float*)d_in[2];   // [N]
    float* out          = (float*)d_out;           // [B, N]

    const int B = in_sizes[0] / NN;                // 2048
    const size_t shmem = (size_t)(NN * STRIDE + 2 * NN) * sizeof(float); // 268,288 B

    // gfx1250 WGP has 320 KB LDS; raise the dynamic-LDS cap past the default.
    hipFuncSetAttribute((const void*)softsort_sinkhorn_kernel,
                        hipFuncAttributeMaxDynamicSharedMemorySize, (int)shmem);

    softsort_sinkhorn_kernel<<<B, NN, shmem, stream>>>(x, x_norm, ref, out);
}
